// BiLSTM_CRF_85100482003375
// MI455X (gfx1250) — matrix operand order repaired and checked by reading, not compiled, not run on hardware
//
#include <hip/hip_runtime.h>
#include <hip/hip_bf16.h>
#include <math.h>

#define V_SZ 100000
#define E_DIM 256
#define H_DIM 512
#define HH 256
#define G4 1024      // 4*HH
#define N2 2048      // both directions fused
#define L_SZ 20
#define T_LEN 4096
#define START_TAG 18
#define STOP_TAG 19
#define NEG_VAL -10000.0f

// recurrence tiling: 64 n-tiles of 16 gate outputs; 38 tiles LDS-resident
#define NTILES 64
#define NT_CACHED 38
#define ROWS_CACHED (NT_CACHED * 16)          // 608 rows
#define WLDS_BYTES (ROWS_CACHED * HH * 2)     // 311296 B
#define SMEM_BYTES (WLDS_BYTES + 2 * G4 * 4 + G4 * 4 + HH * 2)  // 324096 <= 320KB

typedef __attribute__((ext_vector_type(16))) __bf16 v16bf;
typedef __attribute__((ext_vector_type(8)))  float  v8f;

union FragU {
    v16bf v;
    uint4 q[2];
};

// ---- CDNA5 async-to-LDS path (guarded; falls back to plain loads) ----------
#if defined(__gfx1250__) && __has_builtin(__builtin_amdgcn_global_load_async_to_lds_b128) && \
    __has_builtin(__builtin_amdgcn_global_load_async_to_lds_b32) && \
    __has_builtin(__builtin_amdgcn_s_wait_asynccnt)
#define HAVE_ASYNC 1
typedef int v4i_gcc __attribute__((vector_size(16)));
typedef __attribute__((address_space(1))) v4i_gcc* g_v4p;   // global int4*
typedef __attribute__((address_space(3))) v4i_gcc* l_v4p;   // LDS int4*
typedef __attribute__((address_space(1))) int*      g_i32p; // global int*
typedef __attribute__((address_space(3))) int*      l_i32p; // LDS int*
#else
#define HAVE_ASYNC 0
#endif

__device__ __forceinline__ void wait_async_all() {
#if HAVE_ASYNC
    __builtin_amdgcn_s_wait_asynccnt(0);
#endif
}

// ---------------------------------------------------------------------------
// Kernel 1: embedding gather + f32 -> bf16 convert.  xs_bf[t][e]
// ---------------------------------------------------------------------------
__global__ void k_embed(const int* __restrict__ x, const float* __restrict__ emb,
                        __hip_bfloat16* __restrict__ xs_bf) {
    const int t = blockIdx.x;
    const int e = threadIdx.x;
    const int tok = x[t];
    xs_bf[t * E_DIM + e] = __float2bfloat16(emb[(long long)tok * E_DIM + e]);
}

// ---------------------------------------------------------------------------
// Kernel 2: weight prep.  wih_bf [2048][256], whh_bf [2][1024][256],
// fused bias biasz[2048], wout_bf padded [32][512] (rows>=20 zero)
// ---------------------------------------------------------------------------
__global__ void k_prep(const float* __restrict__ w_ih_f, const float* __restrict__ w_ih_b,
                       const float* __restrict__ w_hh_f, const float* __restrict__ w_hh_b,
                       const float* __restrict__ b_ih_f, const float* __restrict__ b_hh_f,
                       const float* __restrict__ b_ih_b, const float* __restrict__ b_hh_b,
                       const float* __restrict__ w_out,
                       __hip_bfloat16* __restrict__ wih_bf,
                       __hip_bfloat16* __restrict__ whh_bf,
                       float* __restrict__ biasz,
                       __hip_bfloat16* __restrict__ wout_bf) {
    const int idx = blockIdx.x * blockDim.x + threadIdx.x;
    if (idx < N2 * E_DIM) {
        const int row = idx / E_DIM, col = idx % E_DIM;
        const float s = (row < G4) ? w_ih_f[row * E_DIM + col]
                                   : w_ih_b[(row - G4) * E_DIM + col];
        wih_bf[idx] = __float2bfloat16(s);
    }
    if (idx < 2 * G4 * HH) {
        const int d = idx >> 18;                   // 1024*256 = 2^18
        const int r = idx & ((G4 * HH) - 1);
        whh_bf[idx] = __float2bfloat16(d ? w_hh_b[r] : w_hh_f[r]);
    }
    if (idx < N2) {
        biasz[idx] = (idx < G4) ? (b_ih_f[idx] + b_hh_f[idx])
                                : (b_ih_b[idx - G4] + b_hh_b[idx - G4]);
    }
    if (idx < 32 * H_DIM) {
        const int r = idx / H_DIM, c = idx % H_DIM;
        wout_bf[idx] = __float2bfloat16(r < L_SZ ? w_out[r * H_DIM + c] : 0.f);
    }
}

// ---------------------------------------------------------------------------
// WMMA 16x16xK bf16 tile accumulator (contiguous k order: lane half h holds k 16h..16h+15 of each 32-step).
// A: [M][K] row-major bf16, B: [N][K] row-major bf16.
// ---------------------------------------------------------------------------
template <int K>
__device__ __forceinline__ v8f wmma_tile(const __hip_bfloat16* __restrict__ arow,
                                         const __hip_bfloat16* __restrict__ brow,
                                         int half, v8f acc) {
#pragma unroll
    for (int k = 0; k < K; k += 32) {
        FragU a, b;
        a.q[0] = *(const uint4*)(arow + k + half * 16);
        a.q[1] = *(const uint4*)(arow + k + half * 16 + 8);
        b.q[0] = *(const uint4*)(brow + k + half * 16);
        b.q[1] = *(const uint4*)(brow + k + half * 16 + 8);
        acc = __builtin_amdgcn_wmma_f32_16x16x32_bf16(false, a.v, false, b.v,
                                                      (short)0, acc, false, false);
    }
    return acc;
}

// ---------------------------------------------------------------------------
// Kernel 3: pre[t][n] = xs[t] . wih[n] + biasz[n]  (M=4096, N=2048, K=256)
// ---------------------------------------------------------------------------
__global__ void k_pre_gemm(const __hip_bfloat16* __restrict__ xs_bf,
                           const __hip_bfloat16* __restrict__ wih_bf,
                           const float* __restrict__ biasz,
                           float* __restrict__ pre) {
    const int lane = threadIdx.x & 31;
    const int wave = threadIdx.x >> 5;
    const int m0 = blockIdx.x * 16;
    const int n0 = (blockIdx.y * 8 + wave) * 16;
    const int half = lane >> 4;
    const int li = lane & 15;

    const __hip_bfloat16* arow = xs_bf + (m0 + li) * E_DIM;
    const __hip_bfloat16* brow = wih_bf + (n0 + li) * E_DIM;

    v8f acc = {};
    acc = wmma_tile<E_DIM>(arow, brow, half, acc);

    const int col = n0 + li;
    const float bias = biasz[col];
#pragma unroll
    for (int v = 0; v < 8; ++v) {
        const int m = m0 + v + half * 8;
        pre[m * N2 + col] = acc[v] + bias;
    }
}

// ---------------------------------------------------------------------------
// Recurrence matmul helper: wave computes two 16-col gate tiles via WMMA.
// All 16 A rows replicate h (bf16 in LDS) -> every D row holds z.
// ---------------------------------------------------------------------------
__device__ __forceinline__ void rec_mm(const unsigned short* __restrict__ b0base,
                                       const unsigned short* __restrict__ b1base,
                                       const unsigned short* __restrict__ h_bf,
                                       int half, v8f& a0, v8f& a1) {
#pragma unroll
    for (int k = 0; k < HH; k += 32) {
        FragU af, b0, b1;
        af.q[0] = *(const uint4*)(h_bf + k + half * 16);
        af.q[1] = *(const uint4*)(h_bf + k + half * 16 + 8);
        b0.q[0] = *(const uint4*)(b0base + k + half * 16);
        b0.q[1] = *(const uint4*)(b0base + k + half * 16 + 8);
        b1.q[0] = *(const uint4*)(b1base + k + half * 16);
        b1.q[1] = *(const uint4*)(b1base + k + half * 16 + 8);
        a0 = __builtin_amdgcn_wmma_f32_16x16x32_bf16(false, af.v, false, b0.v,
                                                     (short)0, a0, false, false);
        a1 = __builtin_amdgcn_wmma_f32_16x16x32_bf16(false, af.v, false, b1.v,
                                                     (short)0, a1, false, false);
    }
}

// ---------------------------------------------------------------------------
// Kernel 4: sequential LSTM recurrence.  grid 2 (dir), block 1024 (32 waves).
// - 38/64 w_hh tiles cached in LDS (304KB), rest streamed from L2 (bf16)
// - h@w_hh via WMMA (h broadcast to all A rows)
// - pre rows double-buffered in LDS via async-to-LDS loads
// - c in registers, h in LDS (bf16)
// ---------------------------------------------------------------------------
__global__ void __launch_bounds__(1024)
k_lstm(const float* __restrict__ pre,
       const __hip_bfloat16* __restrict__ whh_bf,
       const float* __restrict__ h0, const float* __restrict__ c0,
       __hip_bfloat16* __restrict__ lstm_bf) {
    extern __shared__ char smem[];
    unsigned short* wlds   = (unsigned short*)smem;                       // [608*256]
    float*          prebuf = (float*)(smem + WLDS_BYTES);                 // [2][1024]
    float*          zb     = prebuf + 2 * G4;                             // [1024]
    unsigned short* h_bf   = (unsigned short*)(zb + G4);                  // [256]

    const int dir  = blockIdx.x;
    const int j    = threadIdx.x;
    const int lane = j & 31;
    const int wave = j >> 5;
    const int li   = lane & 15;
    const int half = lane >> 4;

    const unsigned short* wsrc =
        (const unsigned short*)(whh_bf + (size_t)dir * G4 * HH);

    // ---- preload cached weight tiles into LDS (async bulk copy) ----
    {
        const int nq = ROWS_CACHED * HH / 8;   // uint4 chunks = 19456
        for (int i = j; i < nq; i += 1024) {
#if HAVE_ASYNC
            __builtin_amdgcn_global_load_async_to_lds_b128(
                (g_v4p)(const void*)(wsrc + i * 8),
                (l_v4p)(void*)(wlds + i * 8), 0, 0);
#else
            ((uint4*)wlds)[i] = ((const uint4*)wsrc)[i];
#endif
        }
    }

    // ---- init h (bf16, LDS), c (register), and first pre row ----
    float creg = 0.f;
    if (j < HH) {
        const __hip_bfloat16 hb = __float2bfloat16(h0[dir * HH + j]);
        h_bf[j] = *(const unsigned short*)&hb;
        creg = c0[dir * HH + j];
    }
    {
        const int t0 = dir ? (T_LEN - 1) : 0;
#if HAVE_ASYNC
        __builtin_amdgcn_global_load_async_to_lds_b32(
            (g_i32p)(const void*)(pre + (size_t)t0 * N2 + dir * G4 + j),
            (l_i32p)(void*)(prebuf + j), 0, 0);
#else
        prebuf[j] = pre[(size_t)t0 * N2 + dir * G4 + j];
#endif
    }

    const bool cached = (2 * wave) < NT_CACHED;   // both of this wave's tiles
    const int r0 = (2 * wave) * 16 + li;          // B row for tile 0
    const int r1 = (2 * wave + 1) * 16 + li;      // B row for tile 1

    for (int s = 0; s < T_LEN; ++s) {
        const int t   = dir ? (T_LEN - 1 - s) : s;
        const int cur = s & 1;

        wait_async_all();        // pre row (and, at s=0, weight preload) resident
        __syncthreads();         // also orders h_bf/zb reuse across steps

        if (s + 1 < T_LEN) {     // kick off next pre row into the other buffer
            const int tn = dir ? (t - 1) : (t + 1);
#if HAVE_ASYNC
            __builtin_amdgcn_global_load_async_to_lds_b32(
                (g_i32p)(const void*)(pre + (size_t)tn * N2 + dir * G4 + j),
                (l_i32p)(void*)(prebuf + (cur ^ 1) * G4 + j), 0, 0);
#else
            prebuf[(cur ^ 1) * G4 + j] = pre[(size_t)tn * N2 + dir * G4 + j];
#endif
        }

        // ---- z = h @ w_hh^T via WMMA (two 16-col tiles per wave) ----
        v8f a0 = {}, a1 = {};
        if (cached) {
            rec_mm(wlds + r0 * HH, wlds + r1 * HH, h_bf, half, a0, a1);
        } else {
            rec_mm(wsrc + r0 * HH, wsrc + r1 * HH, h_bf, half, a0, a1);
        }
        // all D rows identical: lanes 0-15 emit tile0 cols, lanes 16-31 tile1
        zb[wave * 32 + lane] = half ? a1[0] : a0[0];
        __syncthreads();

        // ---- gates + state update (thread j < 256 owns hidden unit j) ----
        if (j < HH) {
            const float* prow = prebuf + cur * G4;
            const float zi = zb[j]          + prow[j];
            const float zf = zb[HH + j]     + prow[HH + j];
            const float zg = zb[2 * HH + j] + prow[2 * HH + j];
            const float zo = zb[3 * HH + j] + prow[3 * HH + j];
            const float ig = 1.f / (1.f + expf(-zi));
            const float fg = 1.f / (1.f + expf(-zf));
            const float gg = tanhf(zg);
            const float og = 1.f / (1.f + expf(-zo));
            creg = fg * creg + ig * gg;
            const float hn = og * tanhf(creg);
            const __hip_bfloat16 hb = __float2bfloat16(hn);
            h_bf[j] = *(const unsigned short*)&hb;
            lstm_bf[(size_t)t * H_DIM + dir * HH + j] = hb;
        }
        __syncthreads();
    }
}

// ---------------------------------------------------------------------------
// Kernel 5: feats[t][l] = lstm_out[t] . w_out[l] + b_out[l]
// M=4096, N=32 (padded from 20), K=512.
// ---------------------------------------------------------------------------
__global__ void k_feat_gemm(const __hip_bfloat16* __restrict__ lstm_bf,
                            const __hip_bfloat16* __restrict__ wout_bf,
                            const float* __restrict__ b_out,
                            float* __restrict__ feats) {
    const int lane = threadIdx.x & 31;
    const int wave = threadIdx.x >> 5;
    const int m0 = blockIdx.x * 16;
    const int n0 = wave * 16;
    const int half = lane >> 4;
    const int li = lane & 15;

    const __hip_bfloat16* arow = lstm_bf + (m0 + li) * H_DIM;
    const __hip_bfloat16* brow = wout_bf + (n0 + li) * H_DIM;

    v8f acc = {};
    acc = wmma_tile<H_DIM>(arow, brow, half, acc);

    const int col = n0 + li;
    const float bias = (col < L_SZ) ? b_out[col] : 0.f;
#pragma unroll
    for (int v = 0; v < 8; ++v) {
        const int m = m0 + v + half * 8;
        feats[m * 32 + col] = acc[v] + bias;
    }
}

// ---------------------------------------------------------------------------
// Kernel 6: Viterbi forward + backtrack.  One wave.
// ---------------------------------------------------------------------------
__global__ void k_viterbi(const float* __restrict__ feats,
                          const float* __restrict__ trans,
                          int* __restrict__ bptr,
                          float* __restrict__ out) {
    __shared__ float tr[L_SZ * L_SZ];
    __shared__ float a[L_SZ];
    const int lane = threadIdx.x;

    for (int i = lane; i < L_SZ * L_SZ; i += 32) tr[i] = trans[i];
    if (lane < L_SZ) a[lane] = (lane == START_TAG) ? 0.f : NEG_VAL;
    __syncthreads();

    for (int t = 0; t < T_LEN; ++t) {
        float na = 0.f;
        if (lane < L_SZ) {
            float best = -3.4e38f;
            int arg = 0;
            for (int from = 0; from < L_SZ; ++from) {
                const float s = a[from] + tr[lane * L_SZ + from];
                if (s > best) { best = s; arg = from; }
            }
            na = best + feats[t * 32 + lane];
            bptr[t * L_SZ + lane] = arg;
        }
        __syncthreads();
        if (lane < L_SZ) a[lane] = na;
        __syncthreads();
    }

    if (lane == 0) {
        float bs = -3.4e38f;
        int bl = 0;
        for (int l = 0; l < L_SZ; ++l) {
            const float s = a[l] + tr[STOP_TAG * L_SZ + l];
            if (s > bs) { bs = s; bl = l; }
        }
        out[0] = bs;
        int cur = bl;
        out[1 + T_LEN - 1] = (float)bl;
        for (int t = T_LEN - 1; t >= 1; --t) {
            const int prev = bptr[t * L_SZ + cur];
            out[1 + t - 1] = (float)prev;
            cur = prev;
        }
    }
}

// ---------------------------------------------------------------------------
// Host launch
// ---------------------------------------------------------------------------
static inline char* carve(char*& p, size_t bytes) {
    char* r = p;
    p += (bytes + 255) & ~size_t(255);
    return r;
}

extern "C" void kernel_launch(void* const* d_in, const int* in_sizes, int n_in,
                              void* d_out, int out_size, void* d_ws, size_t ws_size,
                              hipStream_t stream) {
    const int*   x      = (const int*)  d_in[0];
    const float* emb    = (const float*)d_in[1];
    const float* w_ih_f = (const float*)d_in[2];
    const float* w_hh_f = (const float*)d_in[3];
    const float* b_ih_f = (const float*)d_in[4];
    const float* b_hh_f = (const float*)d_in[5];
    const float* w_ih_b = (const float*)d_in[6];
    const float* w_hh_b = (const float*)d_in[7];
    const float* b_ih_b = (const float*)d_in[8];
    const float* b_hh_b = (const float*)d_in[9];
    const float* w_out  = (const float*)d_in[10];
    const float* b_out  = (const float*)d_in[11];
    const float* trans  = (const float*)d_in[12];
    const float* h0     = (const float*)d_in[13];
    const float* c0     = (const float*)d_in[14];

    char* p = (char*)d_ws;
    __hip_bfloat16* xs_bf   = (__hip_bfloat16*)carve(p, (size_t)T_LEN * E_DIM * 2);
    __hip_bfloat16* wih_bf  = (__hip_bfloat16*)carve(p, (size_t)N2 * E_DIM * 2);
    __hip_bfloat16* whh_bf  = (__hip_bfloat16*)carve(p, (size_t)2 * G4 * HH * 2);
    float*          biasz   = (float*)         carve(p, (size_t)N2 * 4);
    float*          pre     = (float*)         carve(p, (size_t)T_LEN * N2 * 4);
    __hip_bfloat16* lstm_bf = (__hip_bfloat16*)carve(p, (size_t)T_LEN * H_DIM * 2);
    __hip_bfloat16* wout_bf = (__hip_bfloat16*)carve(p, (size_t)32 * H_DIM * 2);
    float*          feats   = (float*)         carve(p, (size_t)T_LEN * 32 * 4);
    int*            bptr    = (int*)           carve(p, (size_t)T_LEN * L_SZ * 4);

    k_embed<<<dim3(T_LEN), dim3(E_DIM), 0, stream>>>(x, emb, xs_bf);

    k_prep<<<dim3((2 * G4 * HH + 255) / 256), dim3(256), 0, stream>>>(
        w_ih_f, w_ih_b, w_hh_f, w_hh_b, b_ih_f, b_hh_f, b_ih_b, b_hh_b, w_out,
        wih_bf, whh_bf, biasz, wout_bf);

    k_pre_gemm<<<dim3(T_LEN / 16, N2 / 128), dim3(256), 0, stream>>>(
        xs_bf, wih_bf, biasz, pre);

    k_lstm<<<dim3(2), dim3(1024), SMEM_BYTES, stream>>>(
        pre, whh_bf, h0, c0, lstm_bf);

    k_feat_gemm<<<dim3(T_LEN / 16), dim3(64), 0, stream>>>(
        lstm_bf, wout_bf, b_out, feats);

    k_viterbi<<<dim3(1), dim3(32), 0, stream>>>(feats, trans, bptr, (float*)d_out);
}
